// GNN3DeepMultiHead_9259949491029
// MI455X (gfx1250) — matrix operand order repaired and checked by reading, compiled, not run on hardware
//
#include <hip/hip_runtime.h>
#include <hip/hip_bf16.h>

// ---------------------------------------------------------------------------
// GNN message-passing network on gfx1250 (CDNA5, wave32, WMMA + async-to-LDS).
//
// m is kept ONLY as f16 ("m_h"), rows padded to 40 halves (80B = 5 x 16B) so
// every gathered row is a clean sequence of global_load_async_to_lds_b128.
// K-layout of the edge GEMM:  [ m[col](36) pad(4) | m[row](36) pad(4) | 0(16) ]
// K-layout of the node GEMM:  [ mi(36) | mo(36) | m_h(40 incl pad) | 0(16) ]
// Pads are annihilated by zero rows in the fragment-packed weights.
// ---------------------------------------------------------------------------
#define N_NODES  50000
#define N_EDGES  400000
#define IN_DIM   4
#define HID      32
#define NCLS     4
#define N_ITERS  3
#define HI       (HID + IN_DIM)      // 36 live feats of m
#define MH       40                  // padded m_h row (halves), 80 bytes
#define NROWS    (N_NODES * NCLS)    // 200000 node-class rows
#define EROWS    (N_EDGES * NCLS)    // 1600000 edge-class rows
#define EKP      96                  // edge GEMM K pad (3 x 32)
#define NKP      128                 // node GEMM K pad (4 x 32)

typedef __attribute__((ext_vector_type(16))) _Float16 v16h;
typedef __attribute__((ext_vector_type(8)))  _Float16 h8;
typedef __attribute__((ext_vector_type(4)))  _Float16 h4;
typedef __attribute__((ext_vector_type(8)))  float    v8f;

// Branch-free tanh: 1 - 2/(exp(2x)+1) via v_exp_f32 + v_rcp_f32.
// Saturates to +/-1 for large |x|, propagates NaN, no EXEC manipulation.
__device__ __forceinline__ float fast_tanh(float x) {
    float e = __builtin_amdgcn_exp2f(x * 2.88539008177793f);   // exp(2x)
    return 1.0f - 2.0f * __builtin_amdgcn_rcpf(e + 1.0f);
}
__device__ __forceinline__ float fast_exp(float x) {
    return __builtin_amdgcn_exp2f(x * 1.44269504088896f);
}

// A-fragment (contiguous K order used here, matched by the packed B): lane<16 holds row=lane, K in
// {0..15}; lane>=16 holds row=lane-16, K in {16..31}.
__device__ __forceinline__ v16h load_afrag(const _Float16* p0, const _Float16* p1) {
    h8 lo = *(const h8*)p0;
    h8 hi = *(const h8*)p1;
    v16h r;
#pragma unroll
    for (int i = 0; i < 8; ++i) { r[i] = lo[i]; r[i + 8] = hi[i]; }
    return r;
}
// B-fragment: 16 contiguous halves (pre-packed fragment order).
__device__ __forceinline__ v16h load_bfrag(const _Float16* p) {
    h8 lo = *(const h8*)p;
    h8 hi = *(const h8*)(p + 8);
    v16h r;
#pragma unroll
    for (int i = 0; i < 8; ++i) { r[i] = lo[i]; r[i + 8] = hi[i]; }
    return r;
}
__device__ __forceinline__ v8f wmma16(v16h a, v16h b, v8f c) {
    return __builtin_amdgcn_wmma_f32_16x16x32_f16(false, a, false, b, (short)0, c,
                                                  false, false);
}

// CDNA5 async global->LDS copy of one padded m_h row (80 bytes, 5 x b128).
// INST_OFFSET applies to BOTH the LDS and the global address (ISA 15.18.3).
__device__ __forceinline__ void async_copy_row80(unsigned lds_off,
                                                 const _Float16* gsrc) {
    asm volatile(
        "global_load_async_to_lds_b128 %0, %1, off\n\t"
        "global_load_async_to_lds_b128 %0, %1, off offset:16\n\t"
        "global_load_async_to_lds_b128 %0, %1, off offset:32\n\t"
        "global_load_async_to_lds_b128 %0, %1, off offset:48\n\t"
        "global_load_async_to_lds_b128 %0, %1, off offset:64"
        :: "v"(lds_off), "v"(gsrc) : "memory");
}
__device__ __forceinline__ void async_wait0() {
    asm volatile("s_wait_asynccnt 0x0" ::: "memory");
}
__device__ __forceinline__ unsigned lds_offset_of(const void* p) {
    return (unsigned)(unsigned long long)p;   // low 32 bits = LDS byte offset
}

// fp32x4 -> 4 packed halves, one ds_store_b64.
__device__ __forceinline__ void store_h4(_Float16* dst, float4 v) {
    h4 t;
    t[0] = (_Float16)v.x; t[1] = (_Float16)v.y;
    t[2] = (_Float16)v.z; t[3] = (_Float16)v.w;
    *(h4*)dst = t;
}

// ---------------------------------------------------------------------------
// Weight packing into WMMA B-fragment order with the padded K maps.
// Fragment layout: [ks][nt][lane][i]; K = ks*32 + (lane>=16?16:0) + i,
// N = nt*16 + (lane&15).
// ---------------------------------------------------------------------------
__global__ void prep_weights_kernel(const float* __restrict__ eW1,
                                    const float* __restrict__ nW1,
                                    const float* __restrict__ nW2,
                                    _Float16* __restrict__ eW1p,
                                    _Float16* __restrict__ nW1p,
                                    _Float16* __restrict__ nW2p) {
    int idx = blockIdx.x * blockDim.x + threadIdx.x;
    if (idx >= 3 * 512) return;
    int j = idx / 512;
    int t = idx % 512;
    const float* eW1j = eW1 + j * 72 * HID;
    const float* nW1j = nW1 + j * 108 * HID;
    const float* nW2j = nW2 + j * HID * HID;
    _Float16* eo  = eW1p + j * (3 * 2 * 32 * 16);
    _Float16* n1o = nW1p + j * (4 * 2 * 32 * 16);
    _Float16* n2o = nW2p + j * (1 * 2 * 32 * 16);
    if (t < 192) {                       // eW1: K map 0..35 -> rows 0..35,
        int lane = t & 31, nt = (t >> 5) & 1, ks = t >> 6;   // 40..75 -> 36..71
        int n = nt * 16 + (lane & 15);
        int kb = ks * 32 + ((lane >> 4) * 16);
#pragma unroll
        for (int i = 0; i < 16; ++i) {
            int K = kb + i;
            int src = (K < 36) ? K : ((K >= 40 && K < 76) ? K - 4 : -1);
            eo[t * 16 + i] = (_Float16)(src >= 0 ? eW1j[src * HID + n] : 0.0f);
        }
    } else if (t < 448) {                // nW1: K<108 direct, else 0
        int t2 = t - 192;
        int lane = t2 & 31, nt = (t2 >> 5) & 1, ks = t2 >> 6;
        int n = nt * 16 + (lane & 15);
        int kb = ks * 32 + ((lane >> 4) * 16);
#pragma unroll
        for (int i = 0; i < 16; ++i) {
            int K = kb + i;
            n1o[t2 * 16 + i] = (_Float16)((K < 108) ? nW1j[K * HID + n] : 0.0f);
        }
    } else {                             // nW2: K<32
        int t2 = t - 448;
        int lane = t2 & 31, nt = t2 >> 5;
        int n = nt * 16 + (lane & 15);
        int kb = (lane >> 4) * 16;
#pragma unroll
        for (int i = 0; i < 16; ++i)
            n2o[t2 * 16 + i] = (_Float16)nW2j[(kb + i) * HID + n];
    }
}

// ---------------------------------------------------------------------------
// Init: m_h[n,c] = [ tanh(x@inW+inb)(32) | x(4) | 0(4) ]  (f16, padded row)
// ---------------------------------------------------------------------------
__global__ void init_kernel(const float* __restrict__ x,
                            const float* __restrict__ inW,
                            const float* __restrict__ inb,
                            _Float16* __restrict__ mh) {
    int idx = blockIdx.x * blockDim.x + threadIdx.x;
    if (idx >= NROWS) return;
    int n = idx >> 2;
    float xv[IN_DIM];
#pragma unroll
    for (int i = 0; i < IN_DIM; ++i) xv[i] = x[n * IN_DIM + i];
    _Float16* dst = mh + (size_t)idx * MH;
#pragma unroll 4
    for (int h = 0; h < HID; ++h) {
        float a = inb[h];
#pragma unroll
        for (int i = 0; i < IN_DIM; ++i) a += xv[i] * inW[i * HID + h];
        dst[h] = (_Float16)fast_tanh(a);
    }
#pragma unroll
    for (int i = 0; i < IN_DIM; ++i) dst[HID + i] = (_Float16)xv[i];
#pragma unroll
    for (int i = 0; i < 4; ++i) dst[HI + i] = (_Float16)0.0f;
}

// ---------------------------------------------------------------------------
// Edge network. Per wave: 16 edge-class rows (4 edges x 4 classes).
// Gather = 10 async b128 copies/row-pair straight into the LDS A-tile.
// ---------------------------------------------------------------------------
__global__ __launch_bounds__(128) void edge_kernel(
    const _Float16* __restrict__ mh,
    const int* __restrict__ rowI, const int* __restrict__ colI,
    const _Float16* __restrict__ eW1p,
    const float* __restrict__ eb1, const float* __restrict__ eW2,
    const float* __restrict__ eb2, float* __restrict__ eaOut) {
    __shared__ __align__(16) _Float16 sA[4][16][EKP];   // 12 KB
    __shared__ __align__(16) float    sT[4][16][HID];   // 8 KB
    __shared__ __align__(16) float    sT2[4][16];

    const int w  = threadIdx.x >> 5;
    const int l  = threadIdx.x & 31;
    const int r  = l & 15;
    const int p  = l >> 4;          // also the A-fragment K-half selector
    const int g0 = (blockIdx.x * 4 + w) * 16;

    // zero pad K 80..95 (16 halves per row, one h8 store per lane)
    {
        h8 z = {};
        *(h8*)&sA[w][r][80 + p * 8] = z;
    }
    // async gather: p==0 -> m_h[col] into k0..39, p==1 -> m_h[row] into k40..79
    {
        int g = g0 + r;
        int e = g >> 2, c = g & 3;
        int node = (p == 0) ? colI[e] : rowI[e];
        const _Float16* src = mh + (size_t)(node * NCLS + c) * MH;
        async_copy_row80(lds_offset_of(&sA[w][r][p * 40]), src);
    }
    async_wait0();
    __syncthreads();

    // GEMM1: [16 x 96] @ [96 x 32]
    v8f acc0 = {};
    v8f acc1 = {};
#pragma unroll
    for (int ks = 0; ks < 3; ++ks) {
        v16h a = load_afrag(&sA[w][r][ks * 32 + p * 16],
                            &sA[w][r][ks * 32 + 8 + p * 16]);
        v16h b0 = load_bfrag(eW1p + ((ks * 2 + 0) * 32 + l) * 16);
        v16h b1 = load_bfrag(eW1p + ((ks * 2 + 1) * 32 + l) * 16);
        acc0 = wmma16(a, b0, acc0);
        acc1 = wmma16(a, b1, acc1);
    }

    // t1 = tanh(acc + eb1) -> LDS (C layout: M = v + 8*p, N = (l&15) + 16*nt)
    {
        int n0 = l & 15;
        float bb0 = eb1[n0], bb1 = eb1[n0 + 16];
#pragma unroll
        for (int v = 0; v < 8; ++v) {
            int M = v + 8 * p;
            sT[w][M][n0]      = fast_tanh(acc0[v] + bb0);
            sT[w][M][n0 + 16] = fast_tanh(acc1[v] + bb1);
        }
    }
    __syncthreads();

    // GEMM2 (32 -> 1) + tanh, vectorized float4 LDS reads
    if (l < 16) {
        const float4* tp = (const float4*)&sT[w][l][0];
        const float4* wp = (const float4*)eW2;
        float s = eb2[0];
#pragma unroll
        for (int q = 0; q < 8; ++q) {
            float4 tv = tp[q], wv = wp[q];
            s += tv.x * wv.x + tv.y * wv.y + tv.z * wv.z + tv.w * wv.w;
        }
        sT2[w][l] = fast_tanh(s);
    }
    __syncthreads();

    // softmax over the 4 classes of each edge
    if (l < 16) {
        int base = l & ~3;
        float x0 = sT2[w][base], x1 = sT2[w][base + 1];
        float x2 = sT2[w][base + 2], x3 = sT2[w][base + 3];
        float mx = fmaxf(fmaxf(x0, x1), fmaxf(x2, x3));
        float den = fast_exp(x0 - mx) + fast_exp(x1 - mx) +
                    fast_exp(x2 - mx) + fast_exp(x3 - mx);
        eaOut[g0 + l] = fast_exp(sT2[w][l] - mx) / den;
    }
}

// ---------------------------------------------------------------------------
// Scatter: mi[col] += ea*m[row];  mo[row] += ea*m[col]  (fp32 L2 atomics).
// ---------------------------------------------------------------------------
__global__ void scatter_kernel(const _Float16* __restrict__ mh,
                               const float* __restrict__ ea,
                               const int* __restrict__ rowI,
                               const int* __restrict__ colI,
                               float* __restrict__ mi, float* __restrict__ mo) {
    int idx = blockIdx.x * blockDim.x + threadIdx.x;
    if (idx >= EROWS) return;
    int e = idx >> 2, c = idx & 3;
    float a = ea[idx];
    int rn = rowI[e], cn = colI[e];
    const _Float16* fr = mh + (size_t)(rn * NCLS + c) * MH;
    const _Float16* fc = mh + (size_t)(cn * NCLS + c) * MH;
    float* pmi = mi + (size_t)(cn * NCLS + c) * HI;
    float* pmo = mo + (size_t)(rn * NCLS + c) * HI;
#pragma unroll 4
    for (int k = 0; k < HI; ++k) {
        atomicAdd(pmi + k, a * (float)fr[k]);
        atomicAdd(pmo + k, a * (float)fc[k]);
    }
}

// ---------------------------------------------------------------------------
// Node network: M = [mi, mo, m_h(padded)] K=128.
//   t = tanh(M @ nW1 + nb1); H = tanh(t @ nW2 + nb2) -> m_h[:, 0:32] in place.
// ---------------------------------------------------------------------------
__global__ __launch_bounds__(128) void node_kernel(
    _Float16* __restrict__ mh, const float* __restrict__ mi,
    const float* __restrict__ mo,
    const _Float16* __restrict__ nW1p, const _Float16* __restrict__ nW2p,
    const float* __restrict__ nb1, const float* __restrict__ nb2) {
    __shared__ __align__(16) _Float16 sM[4][16][NKP];   // 16 KB
    __shared__ __align__(16) _Float16 sH[4][16][HID];   // 4 KB

    const int w  = threadIdx.x >> 5;
    const int l  = threadIdx.x & 31;
    const int r  = l & 15;
    const int p  = l >> 4;
    const int g0 = (blockIdx.x * 4 + w) * 16;

    // gather [mi(0..35) | mo(36..71) | m_h(72..111) | 0(112..127)]
    {
        int g = g0 + r;
        size_t off = (size_t)g * HI;
        if (p == 0) {
            const float4* s4 = (const float4*)(mi + off);
#pragma unroll
            for (int q = 0; q < 9; ++q) store_h4(&sM[w][r][q * 4], s4[q]);
            async_copy_row80(lds_offset_of(&sM[w][r][72]),
                             mh + (size_t)g * MH);
        } else {
            const float4* s4 = (const float4*)(mo + off);
#pragma unroll
            for (int q = 0; q < 9; ++q) store_h4(&sM[w][r][36 + q * 4], s4[q]);
            h8 z = {};
            *(h8*)&sM[w][r][112] = z;
            *(h8*)&sM[w][r][120] = z;
        }
    }
    async_wait0();
    __syncthreads();

    // GEMM1: [16 x 128] @ [128 x 32]
    v8f acc0 = {};
    v8f acc1 = {};
#pragma unroll
    for (int ks = 0; ks < 4; ++ks) {
        v16h a = load_afrag(&sM[w][r][ks * 32 + p * 16],
                            &sM[w][r][ks * 32 + 8 + p * 16]);
        v16h b0 = load_bfrag(nW1p + ((ks * 2 + 0) * 32 + l) * 16);
        v16h b1 = load_bfrag(nW1p + ((ks * 2 + 1) * 32 + l) * 16);
        acc0 = wmma16(a, b0, acc0);
        acc1 = wmma16(a, b1, acc1);
    }

    // t = tanh(acc + nb1) -> LDS f16 A-tile for GEMM2
    const int n0 = l & 15;
    {
        float bb0 = nb1[n0], bb1 = nb1[n0 + 16];
#pragma unroll
        for (int v = 0; v < 8; ++v) {
            int M = v + 8 * p;
            sH[w][M][n0]      = (_Float16)fast_tanh(acc0[v] + bb0);
            sH[w][M][n0 + 16] = (_Float16)fast_tanh(acc1[v] + bb1);
        }
    }
    __syncthreads();

    // GEMM2: [16 x 32] @ [32 x 32]
    v8f d0 = {};
    v8f d1 = {};
    {
        v16h a  = load_afrag(&sH[w][r][p * 16], &sH[w][r][8 + p * 16]);
        v16h c0 = load_bfrag(nW2p + (0 * 32 + l) * 16);
        v16h c1 = load_bfrag(nW2p + (1 * 32 + l) * 16);
        d0 = wmma16(a, c0, d0);
        d1 = wmma16(a, c1, d1);
    }

    // H = tanh(d + nb2) written in place into m_h[:, 0:32]
    {
        float bb0 = nb2[n0], bb1 = nb2[n0 + 16];
#pragma unroll
        for (int v = 0; v < 8; ++v) {
            int M = v + 8 * p;
            size_t g = (size_t)(g0 + M) * MH;
            mh[g + n0]      = (_Float16)fast_tanh(d0[v] + bb0);
            mh[g + n0 + 16] = (_Float16)fast_tanh(d1[v] + bb1);
        }
    }
}

// ---------------------------------------------------------------------------
extern "C" void kernel_launch(void* const* d_in, const int* in_sizes, int n_in,
                              void* d_out, int out_size, void* d_ws, size_t ws_size,
                              hipStream_t stream) {
    const float* x    = (const float*)d_in[0];
    const int*   eIdx = (const int*)d_in[1];
    const float* in_W = (const float*)d_in[2];
    const float* in_b = (const float*)d_in[3];
    const float* eW1  = (const float*)d_in[4];
    const float* eb1  = (const float*)d_in[5];
    const float* eW2  = (const float*)d_in[6];
    const float* eb2  = (const float*)d_in[7];
    const float* nW1  = (const float*)d_in[8];
    const float* nb1  = (const float*)d_in[9];
    const float* nW2  = (const float*)d_in[10];
    const float* nb2  = (const float*)d_in[11];
    (void)in_sizes; (void)n_in; (void)out_size; (void)ws_size;

    char* ws = (char*)d_ws;
    size_t off = 0;
    auto carve = [&](size_t bytes) -> void* {
        void* p = ws + off;
        off = (off + bytes + 255) & ~(size_t)255;
        return p;
    };
    _Float16* mh   = (_Float16*)carve((size_t)3 * NROWS * MH * 2);        // 48 MB
    float*    mi   = (float*)carve((size_t)NROWS * HI * sizeof(float));   // 28.8 MB
    float*    mo   = (float*)carve((size_t)NROWS * HI * sizeof(float));   // 28.8 MB
    float*    ea   = (float*)carve((size_t)EROWS * sizeof(float));        // 6.4 MB
    _Float16* eW1p = (_Float16*)carve((size_t)3 * 3 * 2 * 32 * 16 * 2);
    _Float16* nW1p = (_Float16*)carve((size_t)3 * 4 * 2 * 32 * 16 * 2);
    _Float16* nW2p = (_Float16*)carve((size_t)3 * 1 * 2 * 32 * 16 * 2);

    prep_weights_kernel<<<6, 256, 0, stream>>>(eW1, nW1, nW2, eW1p, nW1p, nW2p);

    for (int j = 0; j < 3; ++j) {
        init_kernel<<<(NROWS + 255) / 256, 256, 0, stream>>>(
            x + (size_t)j * N_NODES * IN_DIM, in_W + j * IN_DIM * HID,
            in_b + j * HID, mh + (size_t)j * NROWS * MH);
    }

    const int edgeBlocks = EROWS / 64;   // 25000
    const int nodeBlocks = NROWS / 64;   // 3125
    const int scatBlocks = (EROWS + 255) / 256;

    for (int it = 0; it < N_ITERS; ++it) {
        for (int j = 0; j < 3; ++j) {
            const int* rowI = eIdx + (size_t)j * 2 * N_EDGES;
            const int* colI = rowI + N_EDGES;
            _Float16* mhj = mh + (size_t)j * NROWS * MH;

            edge_kernel<<<edgeBlocks, 128, 0, stream>>>(
                mhj, rowI, colI, eW1p + j * (3 * 2 * 32 * 16),
                eb1 + j * HID, eW2 + j * HID, eb2 + j, ea);

            hipMemsetAsync(mi, 0, (size_t)NROWS * HI * sizeof(float), stream);
            hipMemsetAsync(mo, 0, (size_t)NROWS * HI * sizeof(float), stream);

            scatter_kernel<<<scatBlocks, 256, 0, stream>>>(mhj, ea, rowI, colI,
                                                           mi, mo);

            node_kernel<<<nodeBlocks, 128, 0, stream>>>(
                mhj, mi, mo, nW1p + j * (4 * 2 * 32 * 16),
                nW2p + j * (1 * 2 * 32 * 16), nb1 + j * HID, nb2 + j * HID);
        }
    }

    for (int j = 0; j < 3; ++j) {
        const int* rowI = eIdx + (size_t)j * 2 * N_EDGES;
        const int* colI = rowI + N_EDGES;
        _Float16* mhj = mh + (size_t)j * NROWS * MH;
        edge_kernel<<<edgeBlocks, 128, 0, stream>>>(
            mhj, rowI, colI, eW1p + j * (3 * 2 * 32 * 16),
            eb1 + j * HID, eW2 + j * HID, eb2 + j,
            (float*)d_out + (size_t)j * EROWS);
    }
}
